// Agent_40372692583315
// MI455X (gfx1250) — matrix operand order repaired and checked by reading, not compiled, not run on hardware
//
#include <hip/hip_runtime.h>

namespace {

constexpr int T_STEPS = 4096;
constexpr int B_ENV   = 1024;
constexpr long TBL    = (long)T_STEPS * B_ENV;

typedef __attribute__((ext_vector_type(16))) _Float16 v16h;
typedef __attribute__((ext_vector_type(8)))  float    v8f;

__device__ __forceinline__ float fexp_(float x) {
    return __builtin_amdgcn_exp2f(x * 1.4426950408889634f);
}
__device__ __forceinline__ float fln_(float x) {
    return __builtin_amdgcn_logf(x) * 0.6931471805599453f;
}
__device__ __forceinline__ float frcp_(float x) { return __builtin_amdgcn_rcpf(x); }
__device__ __forceinline__ float fsig_(float x) { return frcp_(1.0f + fexp_(-x)); }
__device__ __forceinline__ float ftanh_(float x) {
    return fmaf(2.0f, frcp_(1.0f + fexp_(-2.0f * x)), -1.0f);
}

} // namespace

__global__ __launch_bounds__(32)
void lstm_ac_fused(const float* __restrict__ x,        // [T*B, 2]
                   const int*   __restrict__ done,     // [T*B]
                   const int*   __restrict__ action,   // [T*B]
                   const float* __restrict__ h0,       // [B, 16]
                   const float* __restrict__ c0,       // [B, 16]
                   const float* __restrict__ w1,       // [16, 2]
                   const float* __restrict__ b1,       // [16]
                   const float* __restrict__ w_ih,     // [64, 16]
                   const float* __restrict__ w_hh,     // [64, 16]
                   const float* __restrict__ b_ih,     // [64]
                   const float* __restrict__ b_hh,     // [64]
                   const float* __restrict__ actor_w,  // [4, 16]
                   const float* __restrict__ actor_b,  // [4]
                   const float* __restrict__ critic_w, // [1, 16]
                   const float* __restrict__ critic_b, // [1]
                   float* __restrict__ out_action,
                   float* __restrict__ out_logp,
                   float* __restrict__ out_ent,
                   float* __restrict__ out_val,
                   float* __restrict__ out_hN,
                   float* __restrict__ out_cN)
{
    __shared__ float lds_h[16 * 20];              // [env][unit], stride 20 (bank-conflict pad)

    const int lane = threadIdx.x;                 // 0..31 (wave32)
    const int e    = lane & 15;                   // A-side env-in-tile / D-side gate lane
    const int hf   = lane >> 4;                   // half of the wave
    const int tile = blockIdx.x;                  // 0..63
    const int env  = tile * 16 + e;               // global env (A side / outputs)
    const int envD = tile * 16 + 8 * hf;          // base env for D-layout elements

    // ---------------- constants resident in registers ----------------
    // B operand: lane holds column N = n*16+e; elements 0..7 carry w_ih[N][8*hf..8*hf+7]
    // (K 8*hf+k), elements 8..15 carry w_hh[N][8*hf..8*hf+7] (K 16+8*hf+k), 2 f16 per VGPR.
    v16h Bm[4];
    #pragma unroll
    for (int n = 0; n < 4; ++n) {
        const float* ri = w_ih + (n * 16 + e) * 16 + 8 * hf;
        const float* rh = w_hh + (n * 16 + e) * 16 + 8 * hf;
        v16h b;
        #pragma unroll
        for (int k = 0; k < 8; ++k) { b[k] = (_Float16)ri[k]; b[8 + k] = (_Float16)rh[k]; }
        Bm[n] = b;
    }
    float bias[4];
    #pragma unroll
    for (int n = 0; n < 4; ++n) { int g = n * 16 + e; bias[n] = b_ih[g] + b_hh[g]; }

    // encoder weights for units j = 8*hf + k
    float w1a[8], w1b[8], b1v[8];
    #pragma unroll
    for (int k = 0; k < 8; ++k) {
        int j = 8 * hf + k;
        w1a[k] = w1[j * 2 + 0]; w1b[k] = w1[j * 2 + 1]; b1v[k] = b1[j];
    }
    // head weights for the same unit slice
    float awv[4][8], cwv[8], abv[4];
    #pragma unroll
    for (int a = 0; a < 4; ++a) {
        #pragma unroll
        for (int k = 0; k < 8; ++k) awv[a][k] = actor_w[a * 16 + 8 * hf + k];
        abv[a] = actor_b[a];
    }
    #pragma unroll
    for (int k = 0; k < 8; ++k) cwv[k] = critic_w[8 * hf + k];
    const float cbv = critic_b[0];

    // ---------------- loop-carried state ----------------
    float hT[8];                                   // h transposed: env e, units 8*hf+k
    #pragma unroll
    for (int k = 0; k < 8; ++k) hT[k] = h0[env * 16 + 8 * hf + k];
    float cD[8];                                   // c in D layout: env envD+r, unit e
    #pragma unroll
    for (int r = 0; r < 8; ++r) cD[r] = c0[(envD + r) * 16 + e];

    // ---------------- prefetch t = 0 ----------------
    float2 xc  = *(const float2*)(x + (long)env * 2);
    int    dc  = done[env];
    int    ac  = action[env];
    int4   dA  = *(const int4*)(done + envD);
    int4   dB  = *(const int4*)(done + envD + 4);

    for (int t = 0; t < T_STEPS; ++t) {
        // prefetch next step's inputs (overlaps the serial compute below)
        float2 xn = make_float2(0.f, 0.f);
        int dn2 = 0, an2 = 0;
        int4 dAn = make_int4(0, 0, 0, 0), dBn = make_int4(0, 0, 0, 0);
        if (t + 1 < T_STEPS) {
            long ni = (long)(t + 1) * B_ENV + env;
            xn  = *(const float2*)(x + ni * 2);
            dn2 = done[ni];
            an2 = action[ni];
            long nd = (long)(t + 1) * B_ENV + envD;
            dAn = *(const int4*)(done + nd);
            dBn = *(const int4*)(done + nd + 4);
        }

        // ---- build A = [ relu(W1 x + b1)(K 0..15) | m * h(K 16..31) ] in f16 ----
        const float m_e = 1.0f - (float)dc;
        v16h Am;
        #pragma unroll
        for (int k = 0; k < 8; ++k) {
            float hv = fmaxf(fmaf(xc.y, w1b[k], fmaf(xc.x, w1a[k], b1v[k])), 0.0f);
            Am[k]     = (_Float16)hv;
            Am[8 + k] = (_Float16)(hT[k] * m_e);
        }

        // ---- gates: 4 x v_wmma_f32_16x16x32_f16, bias pre-splatted into C ----
        v8f acc[4];
        #pragma unroll
        for (int n = 0; n < 4; ++n) {
            v8f c;
            #pragma unroll
            for (int r = 0; r < 8; ++r) c[r] = bias[n];
            acc[n] = __builtin_amdgcn_wmma_f32_16x16x32_f16(
                false, Am, false, Bm[n], (short)0, c, false, false);
        }

        // ---- LSTM cell in D layout ----
        float mD[8];
        mD[0] = 1.0f - (float)dA.x; mD[1] = 1.0f - (float)dA.y;
        mD[2] = 1.0f - (float)dA.z; mD[3] = 1.0f - (float)dA.w;
        mD[4] = 1.0f - (float)dB.x; mD[5] = 1.0f - (float)dB.y;
        mD[6] = 1.0f - (float)dB.z; mD[7] = 1.0f - (float)dB.w;

        float hD[8];
        #pragma unroll
        for (int r = 0; r < 8; ++r) {
            float i_ = fsig_(acc[0][r]);
            float f_ = fsig_(acc[1][r]);
            float g_ = ftanh_(acc[2][r]);
            float o_ = fsig_(acc[3][r]);
            float c_ = fmaf(f_, cD[r] * mD[r], i_ * g_);
            cD[r] = c_;
            hD[r] = o_ * ftanh_(c_);
        }

        // ---- D-layout -> A-layout transpose of h through LDS ----
        #pragma unroll
        for (int r = 0; r < 8; ++r) lds_h[(8 * hf + r) * 20 + e] = hD[r];
        __builtin_amdgcn_wave_barrier();
        asm volatile("s_wait_dscnt 0" ::: "memory");
        float4 h4a = *(const float4*)(lds_h + e * 20 + 8 * hf);
        float4 h4b = *(const float4*)(lds_h + e * 20 + 8 * hf + 4);
        hT[0] = h4a.x; hT[1] = h4a.y; hT[2] = h4a.z; hT[3] = h4a.w;
        hT[4] = h4b.x; hT[5] = h4b.y; hT[6] = h4b.z; hT[7] = h4b.w;

        // ---- heads: partial dots + half-wave reduction (__shfl_xor 16) ----
        float p0 = 0.f, p1 = 0.f, p2 = 0.f, p3 = 0.f, pv = 0.f;
        #pragma unroll
        for (int k = 0; k < 8; ++k) {
            float hv = hT[k];
            p0 = fmaf(hv, awv[0][k], p0);
            p1 = fmaf(hv, awv[1][k], p1);
            p2 = fmaf(hv, awv[2][k], p2);
            p3 = fmaf(hv, awv[3][k], p3);
            pv = fmaf(hv, cwv[k],    pv);
        }
        p0 += __shfl_xor(p0, 16, 32);
        p1 += __shfl_xor(p1, 16, 32);
        p2 += __shfl_xor(p2, 16, 32);
        p3 += __shfl_xor(p3, 16, 32);
        pv += __shfl_xor(pv, 16, 32);

        float l0 = p0 + abv[0], l1 = p1 + abv[1], l2 = p2 + abv[2], l3 = p3 + abv[3];
        float val = pv + cbv;

        float mx = fmaxf(fmaxf(l0, l1), fmaxf(l2, l3));
        float e0 = fexp_(l0 - mx), e1 = fexp_(l1 - mx), e2 = fexp_(l2 - mx), e3 = fexp_(l3 - mx);
        float s  = e0 + e1 + e2 + e3;
        float lse = mx + fln_(s);
        float lp0 = l0 - lse, lp1 = l1 - lse, lp2 = l2 - lse, lp3 = l3 - lse;
        float rs  = frcp_(s);
        float ent = -(e0 * rs * lp0 + e1 * rs * lp1 + e2 * rs * lp2 + e3 * rs * lp3);
        float logp = (ac == 0) ? lp0 : (ac == 1) ? lp1 : (ac == 2) ? lp2 : lp3;

        if (hf == 0) {
            size_t oidx = (size_t)t * B_ENV + env;
            out_action[oidx] = (float)ac;
            out_logp[oidx]   = logp;
            out_ent[oidx]    = ent;
            out_val[oidx]    = val;
        }

        // rotate prefetched inputs
        xc = xn; dc = dn2; ac = an2; dA = dAn; dB = dBn;
    }

    // ---- final states ----
    *(float4*)(out_hN + (size_t)env * 16 + 8 * hf)     = make_float4(hT[0], hT[1], hT[2], hT[3]);
    *(float4*)(out_hN + (size_t)env * 16 + 8 * hf + 4) = make_float4(hT[4], hT[5], hT[6], hT[7]);
    #pragma unroll
    for (int r = 0; r < 8; ++r)
        out_cN[(size_t)(envD + r) * 16 + e] = cD[r];
}

extern "C" void kernel_launch(void* const* d_in, const int* in_sizes, int n_in,
                              void* d_out, int out_size, void* d_ws, size_t ws_size,
                              hipStream_t stream) {
    (void)in_sizes; (void)n_in; (void)d_ws; (void)ws_size; (void)out_size;

    const float* x        = (const float*)d_in[0];
    const int*   done     = (const int*)  d_in[1];
    const int*   action   = (const int*)  d_in[2];
    const float* h0       = (const float*)d_in[3];
    const float* c0       = (const float*)d_in[4];
    const float* w1       = (const float*)d_in[5];
    const float* b1       = (const float*)d_in[6];
    const float* w_ih     = (const float*)d_in[7];
    const float* w_hh     = (const float*)d_in[8];
    const float* b_ih     = (const float*)d_in[9];
    const float* b_hh     = (const float*)d_in[10];
    const float* actor_w  = (const float*)d_in[11];
    const float* actor_b  = (const float*)d_in[12];
    const float* critic_w = (const float*)d_in[13];
    const float* critic_b = (const float*)d_in[14];

    float* out = (float*)d_out;
    float* out_action = out;                 // [T*B]
    float* out_logp   = out + TBL;           // [T*B]
    float* out_ent    = out + 2 * TBL;       // [T*B]
    float* out_val    = out + 3 * TBL;       // [T*B]
    float* out_hN     = out + 4 * TBL;       // [B, 16]
    float* out_cN     = out_hN + (long)B_ENV * 16;  // [B, 16]

    hipLaunchKernelGGL(lstm_ac_fused, dim3(B_ENV / 16), dim3(32), 0, stream,
                       x, done, action, h0, c0, w1, b1, w_ih, w_hh, b_ih, b_hh,
                       actor_w, actor_b, critic_w, critic_b,
                       out_action, out_logp, out_ent, out_val, out_hN, out_cN);
}
